// GraphAttentionLayer_8315056685320
// MI455X (gfx1250) — matrix operand order repaired and checked by reading, not compiled, not run on hardware
//
#include <hip/hip_runtime.h>

typedef __attribute__((ext_vector_type(16))) _Float16 v16h;
typedef __attribute__((ext_vector_type(8)))  float    v8f;
typedef __attribute__((ext_vector_type(4)))  int      v4i;
typedef __attribute__((ext_vector_type(4)))  float    v4f;

#define N_NODES 8192
#define FIN     128
#define FOUT    64
#define ALPHA   0.2f
#define NEGV    (-1000000000000.0f)

// ---------------------------------------------------------------------------
// Kernel 1: h = inp @ W  (8192x128 * 128x64).  Store h in f32 (for s1/s2) and
// in f16 transposed / feature-major (WMMA B operand: contiguous along K=j).
// ---------------------------------------------------------------------------
__global__ void gat_compute_h(const float* __restrict__ inp,
                              const float* __restrict__ W,
                              float* __restrict__ h32,
                              _Float16* __restrict__ hT16) {
  int idx = blockIdx.x * blockDim.x + threadIdx.x;
  int i = idx >> 6;       // node
  int f = idx & 63;       // feature
  const float* row = inp + (size_t)i * FIN;
  float sum = 0.f;
#pragma unroll 8
  for (int k = 0; k < FIN; ++k) sum = fmaf(row[k], W[k * FOUT + f], sum);
  h32[(size_t)i * FOUT + f] = sum;
  hT16[(size_t)f * N_NODES + i] = (_Float16)sum;
}

// ---------------------------------------------------------------------------
// Kernel 2: s1[i] = h[i,:] . a[0:64],  s2[i] = h[i,:] . a[64:128]
// ---------------------------------------------------------------------------
__global__ void gat_compute_s(const float* __restrict__ h32,
                              const float* __restrict__ a,
                              float* __restrict__ s1,
                              float* __restrict__ s2) {
  int i = blockIdx.x * blockDim.x + threadIdx.x;
  const float* hr = h32 + (size_t)i * FOUT;
  float r1 = 0.f, r2 = 0.f;
#pragma unroll
  for (int f = 0; f < FOUT; ++f) {
    float hv = hr[f];
    r1 = fmaf(hv, a[f], r1);
    r2 = fmaf(hv, a[FOUT + f], r2);
  }
  s1[i] = r1;
  s2[i] = r2;
}

// ---------------------------------------------------------------------------
// Kernel 2b: s2max = max_j s2[j].  Since leaky_relu is monotonic,
// leaky_relu(s1_i + s2max) >= e_ij for every j: a valid (tight) softmax
// shift for every row.  This removes online-softmax from the hot loop.
// ---------------------------------------------------------------------------
__global__ void gat_reduce_s2max(const float* __restrict__ s2,
                                 float* __restrict__ s2max) {
  __shared__ float red[256];
  float local = -3.0e38f;
  for (int i = threadIdx.x; i < N_NODES; i += 256)
    local = fmaxf(local, s2[i]);
  red[threadIdx.x] = local;
  __syncthreads();
#pragma unroll
  for (int s = 128; s > 0; s >>= 1) {
    if (threadIdx.x < s)
      red[threadIdx.x] = fmaxf(red[threadIdx.x], red[threadIdx.x + s]);
    __syncthreads();
  }
  if (threadIdx.x == 0) *s2max = red[0];
}

// ---------------------------------------------------------------------------
// Kernel 3: fused masked-softmax attention + att@h + elu.
// Block = 128 threads = 4 wave32; one 16-row tile per block; each wave owns a
// 2048-column K-slice.  Fixed per-row softmax shift => no rescale, no
// cross-lane traffic in the hot loop: NT b128 adj loads, 16 branch-free
// cndmask+exp, f16 pack, and 4x v_wmma_f32_16x16x32_f16 per 32-col chunk.
// ---------------------------------------------------------------------------
__launch_bounds__(128, 2)
__global__ void gat_fused_attention(const int* __restrict__ adj,
                                    const float* __restrict__ s1,
                                    const float* __restrict__ s2,
                                    const float* __restrict__ s2max,
                                    const _Float16* __restrict__ hT16,
                                    float* __restrict__ out) {
  __shared__ float lds_acc[4][16][FOUT + 4];
  __shared__ float lds_l[4][16];

  const int lane    = threadIdx.x & 31;
  const int wave    = threadIdx.x >> 5;
  const int rowbase = blockIdx.x << 4;
  const int r = lane & 15;   // my matrix row within the 16-row tile
  const int g = lane >> 4;   // lane-group (WMMA K interleave half)

  const float my_s1 = s1[rowbase + r];
  // per-row softmax shift: leaky_relu(s1 + max_j s2)  (upper bound on row max)
  const float xm   = my_s1 + *s2max;
  const float my_M = xm > 0.f ? xm : ALPHA * xm;

  const int* arow = adj + (size_t)(rowbase + r) * N_NODES;

  float l = 0.f;             // running row sum (per lane-group partial)
  v8f acc[4] = {};           // 16x64 f32 accumulator: 4 WMMA C/D tiles

  const int cbeg = wave * (N_NODES / 4);
  const int cend = cbeg + (N_NODES / 4);

  // hoist B base pointers (feature-major h in f16)
  const _Float16* bbase0 = hT16 + (size_t)(0 * 16 + r) * N_NODES + g * 16;
  const _Float16* bbase1 = hT16 + (size_t)(1 * 16 + r) * N_NODES + g * 16;
  const _Float16* bbase2 = hT16 + (size_t)(2 * 16 + r) * N_NODES + g * 16;
  const _Float16* bbase3 = hT16 + (size_t)(3 * 16 + r) * N_NODES + g * 16;

  for (int c0 = cbeg; c0 < cend; c0 += 32) {
    // K offsets owned by this lane for the A (=P) tile (contiguous, matches B):
    //   halves 0..7  -> K = c0 + g*16 + t
    //   halves 8..15 -> K = c0 + g*16 + 8 + (t-8)
    const int jb0 = c0 + g * 16;
    const int jb1 = c0 + g * 16 + 8;

    // unconditional speculative prefetch two chunks ahead (OOB tail is safe:
    // speculative prefetch translation failures are silently dropped)
    __builtin_prefetch(arow + jb0 + 64, 0, 1);

    // adjacency: streamed once (256 MB) -> non-temporal to spare L2
    v4i a0 = __builtin_nontemporal_load((const v4i*)(arow + jb0));
    v4i a1 = __builtin_nontemporal_load((const v4i*)(arow + jb0 + 4));
    v4i a2 = __builtin_nontemporal_load((const v4i*)(arow + jb1));
    v4i a3 = __builtin_nontemporal_load((const v4i*)(arow + jb1 + 4));
    v4f q0 = *(const v4f*)(s2 + jb0);
    v4f q1 = *(const v4f*)(s2 + jb0 + 4);
    v4f q2 = *(const v4f*)(s2 + jb1);
    v4f q3 = *(const v4f*)(s2 + jb1 + 4);

    const int av[16]   = {a0[0], a0[1], a0[2], a0[3], a1[0], a1[1], a1[2],
                          a1[3], a2[0], a2[1], a2[2], a2[3], a3[0], a3[1],
                          a3[2], a3[3]};
    const float qv[16] = {q0[0], q0[1], q0[2], q0[3], q1[0], q1[1], q1[2],
                          q1[3], q2[0], q2[1], q2[2], q2[3], q3[0], q3[1],
                          q3[2], q3[3]};

    v16h A;
    float sump = 0.f;
#pragma unroll
    for (int t = 0; t < 16; ++t) {
      float x  = my_s1 + qv[t];
      float lr = x > 0.f ? x : ALPHA * x;          // leaky_relu (v_cndmask)
      // fold mask into the exp argument: exp(NEGV - M) underflows to 0.0,
      // exactly the reference's where(adj>0, e, NEG) -> softmax semantics.
      // Branch-free: the exp executes unconditionally (no EXEC gymnastics).
      float ez = (av[t] > 0 ? lr : NEGV) - my_M;   // v_cndmask + v_sub
      float p  = __expf(ez);                       // v_exp_f32
      sump += p;
      A[t] = (_Float16)p;
    }
    l += sump;

    // acc += P(16x32 f16) @ h[c0:c0+32, n*16:(n+1)*16]
    // B layout: lanes 0-15 hold K=c0..c0+15, lanes 16-31 hold K=c0+16..c0+31,
    // col N = lane&15; hT16 is feature-major -> contiguous 32B per lane.
    {
      v16h B0 = *(const v16h*)(bbase0 + c0);
      v16h B1 = *(const v16h*)(bbase1 + c0);
      v16h B2 = *(const v16h*)(bbase2 + c0);
      v16h B3 = *(const v16h*)(bbase3 + c0);
      acc[0] = __builtin_amdgcn_wmma_f32_16x16x32_f16(
          false, A, false, B0, (short)0, acc[0], false, false);
      acc[1] = __builtin_amdgcn_wmma_f32_16x16x32_f16(
          false, A, false, B1, (short)0, acc[1], false, false);
      acc[2] = __builtin_amdgcn_wmma_f32_16x16x32_f16(
          false, A, false, B2, (short)0, acc[2], false, false);
      acc[3] = __builtin_amdgcn_wmma_f32_16x16x32_f16(
          false, A, false, B3, (short)0, acc[3], false, false);
    }
  }

  // combine the two lane-group partials of l (disjoint K subsets)
  l += __shfl_xor(l, 16, 32);

  // stage per-wave partials (fixed shift => plain sums, no LSE merge)
  if (g == 0) lds_l[wave][r] = l;
#pragma unroll
  for (int n = 0; n < 4; ++n)
#pragma unroll
    for (int v = 0; v < 8; ++v)
      lds_acc[wave][v + 8 * g][n * 16 + r] = acc[n][v];
  __syncthreads();

  // merge 4 K-slices, apply elu, write out
  for (int idx = threadIdx.x; idx < 16 * FOUT; idx += 128) {
    int i = idx >> 6;
    int f = idx & 63;
    float num = 0.f, den = 0.f;
#pragma unroll
    for (int w = 0; w < 4; ++w) {
      num += lds_acc[w][i][f];
      den += lds_l[w][i];
    }
    float val = num / den;
    out[(size_t)(rowbase + i) * FOUT + f] =
        val > 0.f ? val : (__expf(val) - 1.0f);   // elu (alpha=1)
  }
}

// ---------------------------------------------------------------------------
extern "C" void kernel_launch(void* const* d_in, const int* in_sizes, int n_in,
                              void* d_out, int out_size, void* d_ws,
                              size_t ws_size, hipStream_t stream) {
  const float* inp = (const float*)d_in[0];   // (8192,128) f32
  const int*   adj = (const int*)d_in[1];     // (8192,8192) i32
  const float* W   = (const float*)d_in[2];   // (128,64) f32
  const float* a   = (const float*)d_in[3];   // (128,1) f32
  float* out = (float*)d_out;                 // (8192,64) f32

  char* ws = (char*)d_ws;
  float*    h32   = (float*)ws;                                    // 2 MB
  _Float16* hT16  = (_Float16*)(ws + (size_t)N_NODES * FOUT * 4);  // 1 MB
  float*    s1    = (float*)(ws + (size_t)N_NODES * FOUT * 4 +
                                  (size_t)N_NODES * FOUT * 2);     // 32 KB
  float*    s2    = s1 + N_NODES;                                  // 32 KB
  float*    s2max = s2 + N_NODES;                                  // 4 B

  gat_compute_h<<<(N_NODES * FOUT) / 256, 256, 0, stream>>>(inp, W, h32, hT16);
  gat_compute_s<<<N_NODES / 256, 256, 0, stream>>>(h32, a, s1, s2);
  gat_reduce_s2max<<<1, 256, 0, stream>>>(s2, s2max);
  gat_fused_attention<<<N_NODES / 16, 128, 0, stream>>>(adj, s1, s2, s2max,
                                                        hT16, out);
}
